// _MemoryWorldModel_23175643529264
// MI455X (gfx1250) — matrix operand order repaired and checked by reading, not compiled, not run on hardware
//
#include <hip/hip_runtime.h>

typedef __attribute__((ext_vector_type(16))) __bf16 v16bf;
typedef __attribute__((ext_vector_type(8)))  float  v8f;
typedef __attribute__((ext_vector_type(4)))  unsigned int u32x4;
typedef __attribute__((ext_vector_type(8)))  int i32x8;
typedef __attribute__((ext_vector_type(4)))  int i32x4;

#define DEVINL static __device__ __forceinline__

constexpr int ROWS   = 16 * 4096;  // B*S tokens
constexpr int DIN    = 512;        // INPUT_DIM
constexpr int DK     = 256;        // SLOT_DIM
constexpr int MS     = 64;         // MEM_SLOTS
constexpr int WAVES  = 4;
constexpr int NTHR   = WAVES * 32;
constexpr int WG_ROWS = WAVES * 16;
constexpr int CHUNK  = 16384;      // bf16 elements per 32KB weight chunk

// d_ws layout (bf16 elements)
constexpr int WS_ENC  = 0;                    // enc_w  [256][512]
constexpr int WS_DEC  = WS_ENC + DK * DIN;    // dec_w  [512][256]
constexpr int WS_MEM  = WS_DEC + DIN * DK;    // mem    [64][256] row-major
constexpr int WS_MEMT = WS_MEM + MS * DK;     // mem^T  [256][64]
constexpr int WS_TOT  = WS_MEMT + DK * MS;    // 294912 elems (~576KB)

#if __has_builtin(__builtin_amdgcn_tensor_load_to_lds)
#define HAS_TDM 1
#endif

union FragU { v16bf v; uint4 u[2]; };

DEVINL v8f zero8() { v8f z = {0.f,0.f,0.f,0.f,0.f,0.f,0.f,0.f}; return z; }

// Hardware tanh (gfx1250 V_TANH_F32); trailing v_nop covers TRANS hazard.
DEVINL float fast_tanh(float x) {
  float r;
  asm("v_tanh_f32 %0, %1\n\tv_nop" : "=v"(r) : "v"(x));
  return r;
}

// A fragment from row-major bf16 LDS: m = lane&15, k0 = kb*32 + (lane>=16 ? 8 : 0)
DEVINL v16bf a_frag_lds(const __bf16* base, int ld, int m, int k0) {
  const uint4* p0 = reinterpret_cast<const uint4*>(base + m * ld + k0);
  const uint4* p1 = reinterpret_cast<const uint4*>(base + m * ld + k0 + 16);
  FragU f; f.u[0] = p0[0]; f.u[1] = p1[0];
  return f.v;
}

// B fragment (32x16 bf16): column n contiguous over K in LDS.
DEVINL v16bf b_frag_lds(const __bf16* base, int ld, int n, int k0) {
  const uint4* p0 = reinterpret_cast<const uint4*>(base + n * ld + k0);
  const uint4* p1 = reinterpret_cast<const uint4*>(base + n * ld + k0 + 16);
  FragU f; f.u[0] = p0[0]; f.u[1] = p1[0];
  return f.v;
}

// Cooperative contiguous f32 -> bf16 conversion copy (used for the seq stream)
DEVINL void stage_w(const float* __restrict__ g, __bf16* __restrict__ s,
                    int nelem, int tid) {
  for (int i = tid * 8; i < nelem; i += NTHR * 8) {
    const float4* p = reinterpret_cast<const float4*>(g + i);
    float4 a = p[0], b = p[1];
    union { __bf16 h[8]; uint4 u; } t;
    t.h[0]=(__bf16)a.x; t.h[1]=(__bf16)a.y; t.h[2]=(__bf16)a.z; t.h[3]=(__bf16)a.w;
    t.h[4]=(__bf16)b.x; t.h[5]=(__bf16)b.y; t.h[6]=(__bf16)b.z; t.h[7]=(__bf16)b.w;
    *reinterpret_cast<uint4*>(s + i) = t.u;
  }
}

// Fallback cooperative bf16 -> LDS copy
DEVINL void copy_lds_bf16(const __bf16* __restrict__ g, __bf16* __restrict__ s,
                          int nelem, int tid) {
  for (int i = tid * 8; i < nelem; i += NTHR * 8)
    *reinterpret_cast<uint4*>(s + i) = *reinterpret_cast<const uint4*>(g + i);
}

#ifdef HAS_TDM
// 1-D contiguous TDM copy: nelem bf16 elements from gsrc -> LDS byte offset.
DEVINL void tdm_copy_to_lds(const void* gsrc, unsigned lds_off, int nelem) {
  unsigned long long ga = (unsigned long long)(uintptr_t)gsrc;
  u32x4 g0;
  g0.x = 1u;                                                // count=1, user mode
  g0.y = lds_off;                                           // lds_addr
  g0.z = (unsigned)(ga & 0xFFFFFFFFu);                      // global_addr[31:0]
  g0.w = (unsigned)((ga >> 32) & 0x01FFFFFFu) | (2u << 30); // [56:32] + type=2
  i32x8 g1;
  g1[0] = (int)(1u << 16);                                  // data_size = 2B
  g1[1] = (int)(((unsigned)nelem & 0xFFFFu) << 16);         // tensor_dim0[15:0]
  g1[2] = (int)(((unsigned)nelem >> 16) | (1u << 16));      // dim0[31:16], dim1=1
  g1[3] = (int)(((unsigned)nelem & 0xFFFFu) << 16);         // tile_dim0
  g1[4] = 1;                                                // tile_dim1 = 1
  g1[5] = nelem;                                            // tensor_dim0_stride
  g1[6] = 0;
  g1[7] = 0;
  i32x4 gz = {0, 0, 0, 0};
#if defined(__clang_major__) && __clang_major__ >= 23
  i32x8 gz8 = {0, 0, 0, 0, 0, 0, 0, 0};
  __builtin_amdgcn_tensor_load_to_lds(g0, g1, gz, gz, gz8, 0);
#else
  __builtin_amdgcn_tensor_load_to_lds(g0, g1, gz, gz, 0);
#endif
}
#endif

DEVINL unsigned lds_off(const void* p) { return (unsigned)(uintptr_t)p; }

// ---- Prologue: convert all weights to bf16 once into d_ws ----
__global__ __launch_bounds__(256) void prep_weights(
    const float* __restrict__ enc_w, const float* __restrict__ dec_w,
    const float* __restrict__ memf, __bf16* __restrict__ ws)
{
  int i = blockIdx.x * 256 + threadIdx.x;
  if (i < WS_DEC) {
    ws[i] = (__bf16)enc_w[i];
  } else if (i < WS_MEM) {
    ws[i] = (__bf16)dec_w[i - WS_DEC];
  } else if (i < WS_MEMT) {
    ws[i] = (__bf16)memf[i - WS_MEM];
  } else if (i < WS_TOT) {
    int j = i - WS_MEMT; int slot = j >> 8; int n = j & (DK - 1);
    ws[WS_MEMT + n * MS + slot] = (__bf16)memf[j];
  }
}

__global__ __launch_bounds__(NTHR, 1) void mwm_fused(
    const float* __restrict__ seq,
    const float* __restrict__ enc_b, const float* __restrict__ dec_b,
    const __bf16* __restrict__ encw_bf, const __bf16* __restrict__ decw_bf,
    const __bf16* __restrict__ memw_bf, const __bf16* __restrict__ memtw_bf,
    float* __restrict__ reconO, float* __restrict__ attnO, float* __restrict__ memO)
{
  // 160KB arena:
  //  [0,64K):    sSeq bf16 (stage1) -> sMem + sMemT bf16 (stages 2-3)
  //  [64K,96K):  sW0 (ping)   [96K,128K): sW1 (pong)
  //  [128K,160K): per-wave A-fragment staging (8KB x 4)
  __shared__ alignas(16) unsigned char smem[163840];
  __bf16* sSeq  = reinterpret_cast<__bf16*>(smem);
  __bf16* sMem  = reinterpret_cast<__bf16*>(smem);
  __bf16* sMemT = reinterpret_cast<__bf16*>(smem + 32768);
  __bf16* sW0   = reinterpret_cast<__bf16*>(smem + 65536);
  __bf16* sW1   = reinterpret_cast<__bf16*>(smem + 98304);

  const int tid  = threadIdx.x;
  const int lane = tid & 31;
  const int wave = tid >> 5;
  const int nl   = lane & 15;
  const int hi   = lane >> 4;
  const int row0 = blockIdx.x * WG_ROWS + wave * 16;
  __bf16* mysA = reinterpret_cast<__bf16*>(smem + 131072) + wave * (16 * DK);

  // ---- Stage 0: DMA first enc chunk while converting the seq tile to bf16 ----
#ifdef HAS_TDM
  if (wave == 0) tdm_copy_to_lds(encw_bf, lds_off(sW0), CHUNK);
#else
  copy_lds_bf16(encw_bf, sW0, CHUNK, tid);
#endif
  stage_w(seq + (size_t)blockIdx.x * WG_ROWS * DIN, sSeq, WG_ROWS * DIN, tid);
#ifdef HAS_TDM
  if (wave == 0) __builtin_amdgcn_s_wait_tensorcnt(0);
#endif
  __syncthreads();

  // ---- Stage 1: encoded = tanh(seq @ enc_w^T + enc_b), 16x256 per wave ----
  v8f acc[16];
  #pragma unroll
  for (int t = 0; t < 16; ++t) acc[t] = zero8();

  const __bf16* sSeqW = sSeq + wave * 16 * DIN;

  #pragma unroll
  for (int cc = 0; cc < 8; ++cc) {   // 8 chunks of 32 output columns
    __bf16* cur = (cc & 1) ? sW1 : sW0;
    __bf16* nxt = (cc & 1) ? sW0 : sW1;
    if (cc < 7) {
#ifdef HAS_TDM
      if (wave == 0) tdm_copy_to_lds(encw_bf + (cc + 1) * CHUNK, lds_off(nxt), CHUNK);
#else
      copy_lds_bf16(encw_bf + (cc + 1) * CHUNK, nxt, CHUNK, tid);
#endif
    }
    for (int kb = 0; kb < DIN / 32; ++kb) {
      v16bf a = a_frag_lds(sSeqW, DIN, nl, kb * 32 + hi * 8);
      #pragma unroll
      for (int t = 0; t < 2; ++t) {
        v16bf b = b_frag_lds(cur, DIN, t * 16 + nl, kb * 32 + hi * 8);
        acc[cc * 2 + t] = __builtin_amdgcn_wmma_f32_16x16x32_bf16(
            false, a, false, b, (short)0, acc[cc * 2 + t], false, false);
      }
    }
#ifdef HAS_TDM
    if (cc < 7 && wave == 0) __builtin_amdgcn_s_wait_tensorcnt(0);
#endif
    __syncthreads();
  }

  // ---- DMA mem bank (both layouts) + first dec chunk; overlap with tanh ----
#ifdef HAS_TDM
  if (wave == 0) {
    tdm_copy_to_lds(memw_bf,  lds_off(sMem),  MS * DK);
    tdm_copy_to_lds(memtw_bf, lds_off(sMemT), DK * MS);
    tdm_copy_to_lds(decw_bf,  lds_off(sW0),   CHUNK);
  }
#else
  copy_lds_bf16(memw_bf,  sMem,  MS * DK, tid);
  copy_lds_bf16(memtw_bf, sMemT, DK * MS, tid);
  copy_lds_bf16(decw_bf,  sW0,   CHUNK, tid);
#endif

  // bias + hardware tanh, stash encoded (bf16) per wave
  #pragma unroll
  for (int t = 0; t < 16; ++t) {
    float bias = enc_b[t * 16 + nl];
    #pragma unroll
    for (int v = 0; v < 8; ++v) {
      float e = fast_tanh(acc[t][v] + bias);
      mysA[(v + hi * 8) * DK + t * 16 + nl] = (__bf16)e;
    }
  }
#ifdef HAS_TDM
  if (wave == 0) __builtin_amdgcn_s_wait_tensorcnt(1);  // mem copies done; dec may fly
#endif
  __syncthreads();

  // ---- Stage 2: scores = encoded @ mem^T / 16, softmax over 64 slots ----
  v8f sacc[4];
  #pragma unroll
  for (int t = 0; t < 4; ++t) sacc[t] = zero8();
  #pragma unroll
  for (int kb = 0; kb < DK / 32; ++kb) {
    v16bf a = a_frag_lds(mysA, DK, nl, kb * 32 + hi * 8);
    #pragma unroll
    for (int t = 0; t < 4; ++t) {
      v16bf b = b_frag_lds(sMem, DK, t * 16 + nl, kb * 32 + hi * 8);
      sacc[t] = __builtin_amdgcn_wmma_f32_16x16x32_bf16(
          false, a, false, b, (short)0, sacc[t], false, false);
    }
  }

  constexpr float scale = 0.0625f;  // 1/sqrt(256)
  #pragma unroll
  for (int v = 0; v < 8; ++v) {
    float s0 = sacc[0][v] * scale, s1 = sacc[1][v] * scale;
    float s2 = sacc[2][v] * scale, s3 = sacc[3][v] * scale;
    float mx = fmaxf(fmaxf(s0, s1), fmaxf(s2, s3));
    mx = fmaxf(mx, __shfl_xor(mx, 1, 32));
    mx = fmaxf(mx, __shfl_xor(mx, 2, 32));
    mx = fmaxf(mx, __shfl_xor(mx, 4, 32));
    mx = fmaxf(mx, __shfl_xor(mx, 8, 32));
    float e0 = __expf(s0 - mx), e1 = __expf(s1 - mx);
    float e2 = __expf(s2 - mx), e3 = __expf(s3 - mx);
    float sum = e0 + e1 + e2 + e3;
    sum += __shfl_xor(sum, 1, 32);
    sum += __shfl_xor(sum, 2, 32);
    sum += __shfl_xor(sum, 4, 32);
    sum += __shfl_xor(sum, 8, 32);
    float inv = 1.0f / sum;
    e0 *= inv; e1 *= inv; e2 *= inv; e3 *= inv;
    int grow = row0 + v + hi * 8;
    float* ap = attnO + (size_t)grow * MS + nl;
    ap[0] = e0; ap[16] = e1; ap[32] = e2; ap[48] = e3;
    sacc[0][v] = e0; sacc[1][v] = e1; sacc[2][v] = e2; sacc[3][v] = e3;
  }
  __syncthreads();

  // stash attention (bf16) as 16x64 for stage-3 A fragments
  #pragma unroll
  for (int v = 0; v < 8; ++v) {
    #pragma unroll
    for (int t = 0; t < 4; ++t)
      mysA[(v + hi * 8) * MS + t * 16 + nl] = (__bf16)sacc[t][v];
  }
  __syncthreads();

  // ---- Stage 3: memory = attention @ mem  (16x64 @ 64x256) ----
  #pragma unroll
  for (int t = 0; t < 16; ++t) acc[t] = zero8();
  #pragma unroll
  for (int kc = 0; kc < 2; ++kc) {
    v16bf a = a_frag_lds(mysA, MS, nl, kc * 32 + hi * 8);
    #pragma unroll
    for (int t = 0; t < 16; ++t) {
      v16bf b = b_frag_lds(sMemT, MS, t * 16 + nl, kc * 32 + hi * 8);
      acc[t] = __builtin_amdgcn_wmma_f32_16x16x32_bf16(
          false, a, false, b, (short)0, acc[t], false, false);
    }
  }
  __syncthreads();

  // write memory output (f32) + stash bf16 copy for stage-4 A fragments
  #pragma unroll
  for (int t = 0; t < 16; ++t) {
    #pragma unroll
    for (int v = 0; v < 8; ++v) {
      int grow = row0 + v + hi * 8;
      float val = acc[t][v];
      memO[(size_t)grow * DK + t * 16 + nl] = val;
      mysA[(v + hi * 8) * DK + t * 16 + nl] = (__bf16)val;
    }
  }
#ifdef HAS_TDM
  if (wave == 0) __builtin_amdgcn_s_wait_tensorcnt(0);  // dec chunk 0 landed
#endif
  __syncthreads();

  // ---- Stage 4: reconstruction = memory @ dec_w^T + dec_b (TDM ping-pong) ----
  #pragma unroll
  for (int cc = 0; cc < 8; ++cc) {   // 8 chunks of 64 output columns
    __bf16* cur = (cc & 1) ? sW1 : sW0;
    __bf16* nxt = (cc & 1) ? sW0 : sW1;
    if (cc < 7) {
#ifdef HAS_TDM
      if (wave == 0) tdm_copy_to_lds(decw_bf + (cc + 1) * CHUNK, lds_off(nxt), CHUNK);
#else
      copy_lds_bf16(decw_bf + (cc + 1) * CHUNK, nxt, CHUNK, tid);
#endif
    }
    v8f dacc[4];
    #pragma unroll
    for (int t = 0; t < 4; ++t) dacc[t] = zero8();
    for (int kb = 0; kb < DK / 32; ++kb) {
      v16bf a = a_frag_lds(mysA, DK, nl, kb * 32 + hi * 8);
      #pragma unroll
      for (int t = 0; t < 4; ++t) {
        v16bf b = b_frag_lds(cur, DK, t * 16 + nl, kb * 32 + hi * 8);
        dacc[t] = __builtin_amdgcn_wmma_f32_16x16x32_bf16(
            false, a, false, b, (short)0, dacc[t], false, false);
      }
    }
    #pragma unroll
    for (int t = 0; t < 4; ++t) {
      int col = cc * 64 + t * 16 + nl;
      float bias = dec_b[col];
      #pragma unroll
      for (int v = 0; v < 8; ++v) {
        int grow = row0 + v + hi * 8;
        reconO[(size_t)grow * DIN + col] = dacc[t][v] + bias;
      }
    }
#ifdef HAS_TDM
    if (cc < 7 && wave == 0) __builtin_amdgcn_s_wait_tensorcnt(0);
#endif
    __syncthreads();
  }
}

extern "C" void kernel_launch(void* const* d_in, const int* in_sizes, int n_in,
                              void* d_out, int out_size, void* d_ws, size_t ws_size,
                              hipStream_t stream) {
  (void)in_sizes; (void)n_in; (void)out_size; (void)ws_size;
  const float* seq  = (const float*)d_in[0];
  const float* encw = (const float*)d_in[1];
  const float* encb = (const float*)d_in[2];
  const float* memf = (const float*)d_in[3];
  const float* decw = (const float*)d_in[4];
  const float* decb = (const float*)d_in[5];

  float* out   = (float*)d_out;
  float* recon = out;                                   // [ROWS, 512]
  float* attn  = out + (size_t)ROWS * DIN;              // [ROWS, 64]
  float* memo  = attn + (size_t)ROWS * MS;              // [ROWS, 256]

  __bf16* ws = (__bf16*)d_ws;
  prep_weights<<<dim3((WS_TOT + 255) / 256), 256, 0, stream>>>(encw, decw, memf, ws);

  dim3 grid(ROWS / WG_ROWS);
  mwm_fused<<<grid, NTHR, 0, stream>>>(seq, encb, decb,
                                       ws + WS_ENC, ws + WS_DEC,
                                       ws + WS_MEM, ws + WS_MEMT,
                                       recon, attn, memo);
}
